// NystromSelfAttention_32195074851342
// MI455X (gfx1250) — matrix operand order repaired and checked by reading, not compiled, not run on hardware
//
#include <hip/hip_runtime.h>

// ---------------------------------------------------------------------------
// Nystrom self-attention for MI455X (gfx1250, wave32, WMMA bf16).
// B=8, N=4096, D=512, H=8, M=256, DH=64, ITERS=6.
// GEMMs: v_wmma_f32_16x16x32_bf16, f32 accumulate. Tile staging uses the
// CDNA5 async global->LDS path (NT variant) and the Tensor Data Mover
// (NN variant, one D# per K-step issued by wave 0).
// ---------------------------------------------------------------------------

typedef __bf16 bf16_t;
typedef __attribute__((ext_vector_type(16))) __bf16 v16bf;
typedef __attribute__((ext_vector_type(8)))  __bf16 v8bf;
typedef __attribute__((ext_vector_type(8)))  float  v8f;
typedef __attribute__((ext_vector_type(4)))  unsigned int v4u;
typedef __attribute__((ext_vector_type(8)))  int v8i;
typedef __attribute__((ext_vector_type(4)))  int v4i;

#define B_   8
#define N_   4096
#define D_   512
#define H_   8
#define M_   256
#define DH_  64
#define BH_  (B_ * H_)
#define ITERS_ 6

// ---------------------------------------------------------------------------
// Generic batched GEMM:  D = alpha * (A @ B) [+ beta*C | + bias]
//   A: bf16 [M,K] row-major (lda), batch stride sA (elements)
//   B: BTRANS ? bf16 [N,K] row-major (ldb)  (B^T, NT gemm; async-LDS staging)
//            : bf16 [K,N] row-major (ldb)  (NN gemm; A via TDM, B transposed)
//   EPI: 0 none | 1 D += beta*C (bf16, ldc/sC) | 2 D += bias[col] (f32)
// Tile: 128x64x32, 128 threads (4 waves); wave -> 64x32 via 4x2 WMMA.
// ---------------------------------------------------------------------------
template<bool BTRANS, int EPI, bool OUTF32>
__global__ __launch_bounds__(128)
void gemm_wmma_kernel(const bf16_t* __restrict__ A0, const bf16_t* __restrict__ B0,
                      const void* __restrict__ C0, void* __restrict__ D0,
                      int K, int lda, int ldb, int ldc, int ldd,
                      long long sA, long long sB, long long sC, long long sD,
                      float alpha, float beta)
{
    __shared__ __attribute__((aligned(16))) bf16_t As[128][32];
    __shared__ __attribute__((aligned(16))) bf16_t Bs[64][32]; // [n][k]

    const int bz   = blockIdx.z;
    const bf16_t* A  = A0 + (long long)bz * sA + (size_t)blockIdx.x * 128 * lda;
    const bf16_t* Bp = B0 + (long long)bz * sB;

    const int tid  = threadIdx.x;
    const int wave = tid >> 5, lane = tid & 31;
    const int wr   = (wave & 1) * 64;   // wave row offset in tile
    const int wc   = (wave >> 1) * 32;  // wave col offset in tile
    const int l15  = lane & 15, lh = lane >> 4;

    v8f acc[4][2] = {};

    for (int k0 = 0; k0 < K; k0 += 32) {
        if (BTRANS) {
            // ---- async global->LDS staging (ASYNCcnt path) ----
            // A tile 128x32: thread t owns row t (64B = 4 x b128)
            {
                const bf16_t* ap = &A[(size_t)tid * lda + k0];
                unsigned adst = (unsigned)(size_t)&As[tid][0];
                #pragma unroll
                for (int c = 0; c < 4; ++c) {
                    const bf16_t* g = ap + c * 8;
                    unsigned l = adst + c * 16;
                    asm volatile("global_load_async_to_lds_b128 %0, %1, off"
                                 :: "v"(l), "v"(g) : "memory");
                }
            }
            // B tile 64x32 ([N,K] storage): 2 x b128 per thread
            #pragma unroll
            for (int c = 0; c < 2; ++c) {
                int chunk = tid * 2 + c;
                int r = chunk >> 2, col8 = (chunk & 3) * 8;
                const bf16_t* g = &Bp[((size_t)(blockIdx.y * 64 + r)) * ldb + k0 + col8];
                unsigned l = (unsigned)(size_t)&Bs[r][col8];
                asm volatile("global_load_async_to_lds_b128 %0, %1, off"
                             :: "v"(l), "v"(g) : "memory");
            }
            asm volatile("s_wait_asynccnt 0x0" ::: "memory");
        } else {
            // ---- A tile via Tensor Data Mover (TENSORcnt path) ----
            if (wave == 0) {
                unsigned long long ga = (unsigned long long)(const void*)(A + k0);
                unsigned ldsb = (unsigned)(size_t)&As[0][0];
                v4u g0; v8i g1;
                v4i g2 = {0, 0, 0, 0}, g3 = {0, 0, 0, 0};
                v8i g4 = {0, 0, 0, 0, 0, 0, 0, 0};
                // D# group0: count=1 | lds_addr | global_addr[56:0] | type=2
                g0[0] = 1u;
                g0[1] = ldsb;
                g0[2] = (unsigned)(ga & 0xffffffffu);
                g0[3] = (unsigned)((ga >> 32) & 0x1ffffffu) | (2u << 30);
                // D# group1: data_size=2B; tensor 32 x 128; tile 32 x 128; stride0=lda
                g1[0] = 0x10000;               // data_size=1 (2 bytes)
                g1[1] = (int)(32u << 16);      // tensor_dim0 lo16 = 32
                g1[2] = (int)(128u << 16);     // tensor_dim1 lo16 = 128
                g1[3] = (int)(32u << 16);      // tile_dim0 = 32
                g1[4] = 128;                   // tile_dim1 = 128 (tile_dim2 = 0)
                g1[5] = lda;                   // tensor_dim0_stride lo32
                g1[6] = 0;
                g1[7] = 0;                     // tensor_dim1_stride unused
                __builtin_amdgcn_tensor_load_to_lds(g0, g1, g2, g3, g4, 0);
                __builtin_amdgcn_s_wait_tensorcnt((short)0);
            }
            // B tile ([K,N] storage): transpose while staging to LDS
            int kk = tid >> 2, nb = (tid & 3) * 16;
            const bf16_t* bp = &Bp[((size_t)(k0 + kk)) * ldb + blockIdx.y * 64 + nb];
            if (k0 + 32 < K) __builtin_prefetch(bp + 32 * ldb, 0, 1); // global_prefetch_b8
            #pragma unroll
            for (int i = 0; i < 16; ++i) Bs[nb + i][kk] = bp[i];
        }
        __syncthreads();

        // Fragments per ISA layout.
        // A 16x32: lanes 0-15 row m / K base 0, lanes 16-31 row m / K base +8;
        //          elems 0..7 -> K in [8*lh, +8), elems 8..15 -> [16+8*lh, +8)
        // B 32x16: col n, same K mapping as A (elems 0..7 -> [8*lh,+8), 8..15 -> [16+8*lh,+8))
        v16bf af[4], bfm[2];
        #pragma unroll
        for (int i = 0; i < 4; ++i) {
            int m = wr + i * 16 + l15;
            union { v16bf v; v8bf h[2]; } ua;
            ua.h[0] = *(const v8bf*)(&As[m][8 * lh]);
            ua.h[1] = *(const v8bf*)(&As[m][16 + 8 * lh]);
            af[i] = ua.v;
        }
        #pragma unroll
        for (int j = 0; j < 2; ++j) {
            int n = wc + j * 16 + l15;
            union { v16bf v; v8bf h[2]; } ub;
            ub.h[0] = *(const v8bf*)(&Bs[n][8 * lh]);
            ub.h[1] = *(const v8bf*)(&Bs[n][16 + 8 * lh]);
            bfm[j] = ub.v;
        }
        #pragma unroll
        for (int i = 0; i < 4; ++i)
            #pragma unroll
            for (int j = 0; j < 2; ++j)
                acc[i][j] = __builtin_amdgcn_wmma_f32_16x16x32_bf16(
                    false, af[i], false, bfm[j], (short)0, acc[i][j], false, false);
        __syncthreads();
    }

    // Epilogue. C/D layout: VGPR e: lanes 0-15 -> (row=e, col=lane),
    //                               lanes 16-31 -> (row=e+8, col=lane-16)
    const bf16_t* Cin  = (const bf16_t*)C0 + (EPI == 1 ? (long long)bz * sC : 0);
    const float*  bias = (const float*)C0;
    float*  Df = (float*)D0  + (long long)bz * sD;
    bf16_t* Db = (bf16_t*)D0 + (long long)bz * sD;

    #pragma unroll
    for (int i = 0; i < 4; ++i)
        #pragma unroll
        for (int j = 0; j < 2; ++j) {
            int row0 = blockIdx.x * 128 + wr + i * 16 + 8 * lh;
            int col  = blockIdx.y * 64 + wc + j * 16 + l15;
            #pragma unroll
            for (int e = 0; e < 8; ++e) {
                int row = row0 + e;
                float val = alpha * acc[i][j][e];
                if (EPI == 1) val += beta * (float)Cin[(size_t)row * ldc + col];
                if (EPI == 2) val += bias[col];
                if (OUTF32) Df[(size_t)row * ldd + col] = val;
                else        Db[(size_t)row * ldd + col] = (bf16_t)val;
            }
        }
}

// ---------------------------------------------------------------------------
// Elementwise / layout kernels
// ---------------------------------------------------------------------------
__global__ void f32_to_bf16_kernel(const float* __restrict__ in, bf16_t* __restrict__ out,
                                   unsigned long long n) {
    unsigned long long i = (unsigned long long)blockIdx.x * 256 + threadIdx.x;
    if (i < n) out[i] = (bf16_t)in[i];
}

// in [K,N] f32 -> out [N,K] bf16
__global__ void transpose_w_kernel(const float* __restrict__ in, bf16_t* __restrict__ out,
                                   int K, int N) {
    int idx = blockIdx.x * 256 + threadIdx.x;
    if (idx < K * N) {
        int k = idx / N, n = idx % N;
        out[(size_t)n * K + k] = (bf16_t)in[idx];
    }
}

// qkv [B,N,1536] bf16 -> q,k,v [B,H,N,64] bf16 (q scaled by 1/8)
__global__ void split_qkv_kernel(const bf16_t* __restrict__ qkv,
                                 bf16_t* __restrict__ q, bf16_t* __restrict__ k,
                                 bf16_t* __restrict__ v) {
    unsigned long long i = (unsigned long long)blockIdx.x * 256 + threadIdx.x; // B*N*512
    unsigned long long bn = i >> 9; int c = (int)(i & 511);
    int b = (int)(bn >> 12), n = (int)(bn & 4095);
    int h = c >> 6, d = c & 63;
    size_t src = (size_t)bn * (3 * D_);
    size_t dst = (((size_t)(b * H_ + h)) * N_ + n) * DH_ + d;
    q[dst] = (bf16_t)((float)qkv[src + c] * 0.125f);
    k[dst] = qkv[src + D_ + c];
    v[dst] = qkv[src + 2 * D_ + c];
}

// in [BH,4096,64] -> out [BH,256,64], mean over 16 consecutive rows
__global__ void landmark_kernel(const bf16_t* __restrict__ in, bf16_t* __restrict__ out) {
    unsigned long long i = (unsigned long long)blockIdx.x * 256 + threadIdx.x; // BH*256*64
    unsigned long long bh = i >> 14;
    int m = (int)((i >> 6) & 255), d = (int)(i & 63);
    const bf16_t* p = in + (bh * N_ + (size_t)m * 16) * DH_ + d;
    float s = 0.f;
    #pragma unroll
    for (int t = 0; t < 16; ++t) s += (float)p[t * DH_];
    out[i] = (bf16_t)(s * 0.0625f);
}

// row softmax: f32 in -> bf16 out; one block (256 thr) per row of length C
__global__ void softmax_rows_kernel(const float* __restrict__ in, bf16_t* __restrict__ out,
                                    int C) {
    const int row = blockIdx.x, t = threadIdx.x;
    const float* r = in + (size_t)row * C;
    bf16_t* o = out + (size_t)row * C;
    __shared__ float red[256];
    float m = -1e30f;
    for (int c = t; c < C; c += 256) m = fmaxf(m, r[c]);
    red[t] = m; __syncthreads();
    for (int s = 128; s > 0; s >>= 1) { if (t < s) red[t] = fmaxf(red[t], red[t + s]); __syncthreads(); }
    m = red[0]; __syncthreads();
    float sum = 0.f;
    for (int c = t; c < C; c += 256) sum += __expf(r[c] - m);
    red[t] = sum; __syncthreads();
    for (int s = 128; s > 0; s >>= 1) { if (t < s) red[t] += red[t + s]; __syncthreads(); }
    float inv = 1.0f / red[0];
    for (int c = t; c < C; c += 256) o[c] = (bf16_t)(__expf(r[c] - m) * inv);
}

// per-(b,h): scale = 1 / (max row-abs-sum * max col-abs-sum) of S[256,256] (S>=0)
__global__ void pinv_scale_kernel(const bf16_t* __restrict__ S, float* __restrict__ scales) {
    int bh = blockIdx.x, t = threadIdx.x;
    const bf16_t* s = S + (size_t)bh * M_ * M_;
    float rs = 0.f, cs = 0.f;
    for (int j = 0; j < M_; ++j) { rs += (float)s[(size_t)t * M_ + j]; cs += (float)s[(size_t)j * M_ + t]; }
    __shared__ float r1[256], r2[256];
    r1[t] = rs; r2[t] = cs; __syncthreads();
    for (int k = 128; k > 0; k >>= 1) {
        if (t < k) { r1[t] = fmaxf(r1[t], r1[t + k]); r2[t] = fmaxf(r2[t], r2[t + k]); }
        __syncthreads();
    }
    if (t == 0) scales[bh] = 1.0f / (r1[0] * r2[0]);
}

// z0[bh][i][j] = S[bh][j][i] * scales[bh]
__global__ void pinv_z0_kernel(const bf16_t* __restrict__ S, const float* __restrict__ scales,
                               bf16_t* __restrict__ z) {
    int bh = blockIdx.x, i = blockIdx.y, j = threadIdx.x;
    size_t base = (size_t)bh * M_ * M_;
    z[base + (size_t)i * M_ + j] = (bf16_t)((float)S[base + (size_t)j * M_ + i] * scales[bh]);
}

// outh [B,H,N,64] -> xo [B,N,512]
__global__ void merge_heads_kernel(const bf16_t* __restrict__ outh, bf16_t* __restrict__ xo) {
    unsigned long long i = (unsigned long long)blockIdx.x * 256 + threadIdx.x; // B*N*512
    unsigned long long bn = i >> 9; int c = (int)(i & 511);
    int b = (int)(bn >> 12), n = (int)(bn & 4095);
    int h = c >> 6, d = c & 63;
    xo[i] = outh[(((size_t)(b * H_ + h)) * N_ + n) * DH_ + d];
}

// ---------------------------------------------------------------------------
// Host orchestration
// ---------------------------------------------------------------------------
extern "C" void kernel_launch(void* const* d_in, const int* in_sizes, int n_in,
                              void* d_out, int out_size, void* d_ws, size_t ws_size,
                              hipStream_t stream)
{
    (void)in_sizes; (void)n_in; (void)out_size; (void)ws_size;
    const float* x    = (const float*)d_in[0];
    const float* wqkv = (const float*)d_in[1];
    const float* wout = (const float*)d_in[2];
    const float* bout = (const float*)d_in[3];
    float* out = (float*)d_out;

    // ---- workspace arena ----
    char* base = (char*)d_ws; size_t off = 0;
    auto alloc = [&](size_t bytes) -> void* {
        void* p = base + off; off += (bytes + 255) & ~(size_t)255; return p;
    };
    const size_t XE = (size_t)B_ * N_ * D_;          // 16,777,216
    bf16_t* x_bf  = (bf16_t*)alloc(XE * 2);
    bf16_t* wqkvT = (bf16_t*)alloc((size_t)D_ * 3 * D_ * 2);
    bf16_t* woutT = (bf16_t*)alloc((size_t)D_ * D_ * 2);
    char*   R1    = (char*)  alloc(268435456ull);     // qkv / sim f32 / tA (reused)
    bf16_t* q     = (bf16_t*)alloc(XE * 2);           // later reused as outh
    bf16_t* k     = (bf16_t*)alloc(XE * 2);           // later reused as xo
    bf16_t* v     = (bf16_t*)alloc(XE * 2);
    bf16_t* q_l   = (bf16_t*)alloc((size_t)BH_ * M_ * DH_ * 2);
    bf16_t* k_l   = (bf16_t*)alloc((size_t)BH_ * M_ * DH_ * 2);
    bf16_t* S     = (bf16_t*)alloc((size_t)BH_ * M_ * M_ * 2);
    bf16_t* A1    = (bf16_t*)alloc((size_t)BH_ * N_ * M_ * 2);
    bf16_t* A3    = (bf16_t*)alloc((size_t)BH_ * M_ * N_ * 2);
    bf16_t* z     = (bf16_t*)alloc((size_t)BH_ * M_ * M_ * 2);
    bf16_t* znew  = (bf16_t*)alloc((size_t)BH_ * M_ * M_ * 2);
    bf16_t* T1    = (bf16_t*)alloc((size_t)BH_ * M_ * M_ * 2);
    bf16_t* T3    = (bf16_t*)alloc((size_t)BH_ * M_ * M_ * 2);
    bf16_t* T5    = (bf16_t*)alloc((size_t)BH_ * M_ * M_ * 2);
    bf16_t* a3v   = (bf16_t*)alloc((size_t)BH_ * M_ * DH_ * 2);
    float*  scales= (float*) alloc(BH_ * sizeof(float));

    // ---- 0) precision conversion / weight transpose ----
    f32_to_bf16_kernel<<<(unsigned)(XE / 256), 256, 0, stream>>>(x, x_bf, XE);
    transpose_w_kernel<<<(D_ * 3 * D_ + 255) / 256, 256, 0, stream>>>(wqkv, wqkvT, D_, 3 * D_);
    transpose_w_kernel<<<(D_ * D_ + 255) / 256, 256, 0, stream>>>(wout, woutT, D_, D_);

    // ---- 1) qkv = x @ w_qkv  (NT via pre-transposed weights) ----
    bf16_t* qkv = (bf16_t*)R1;
    gemm_wmma_kernel<true, 0, false><<<dim3((B_ * N_) / 128, (3 * D_) / 64, 1), 128, 0, stream>>>(
        x_bf, wqkvT, nullptr, qkv, /*K*/D_, /*lda*/D_, /*ldb*/D_, 0, /*ldd*/3 * D_,
        0, 0, 0, 0, 1.f, 0.f);

    // ---- 2) head split + landmarks ----
    split_qkv_kernel<<<(unsigned)(XE / 256), 256, 0, stream>>>(qkv, q, k, v);
    landmark_kernel<<<(BH_ * M_ * DH_) / 256, 256, 0, stream>>>(q, q_l);
    landmark_kernel<<<(BH_ * M_ * DH_) / 256, 256, 0, stream>>>(k, k_l);

    float* simf = (float*)R1;
    const long long sMM = (long long)M_ * M_;        // 65536
    const long long sNM = (long long)N_ * M_;        // 1,048,576
    const long long sND = (long long)N_ * DH_;       // 262,144
    const long long sMD = (long long)M_ * DH_;       // 16,384

    // ---- 3) sim2 = q_l @ k_l^T -> softmax -> S ----
    gemm_wmma_kernel<true, 0, true><<<dim3(M_ / 128, M_ / 64, BH_), 128, 0, stream>>>(
        q_l, k_l, nullptr, simf, DH_, DH_, DH_, 0, M_, sMD, sMD, 0, sMM, 1.f, 0.f);
    softmax_rows_kernel<<<BH_ * M_, 256, 0, stream>>>(simf, S, M_);

    // ---- 4) Moore-Penrose init + 6 Newton-Schulz iterations ----
    pinv_scale_kernel<<<BH_, 256, 0, stream>>>(S, scales);
    pinv_z0_kernel<<<dim3(BH_, M_), M_, 0, stream>>>(S, scales, z);

    bf16_t* zc = z; bf16_t* zn = znew;
    for (int it = 0; it < ITERS_; ++it) {
        // T1 = S @ z
        gemm_wmma_kernel<false, 0, false><<<dim3(M_ / 128, M_ / 64, BH_), 128, 0, stream>>>(
            S, zc, nullptr, T1, M_, M_, M_, 0, M_, sMM, sMM, 0, sMM, 1.f, 0.f);
        // T3 = T1 @ (7I - T1) = 7*T1 - T1@T1
        gemm_wmma_kernel<false, 1, false><<<dim3(M_ / 128, M_ / 64, BH_), 128, 0, stream>>>(
            T1, T1, T1, T3, M_, M_, M_, M_, M_, sMM, sMM, sMM, sMM, -1.f, 7.f);
        // T5 = T1 @ (15I - T3) = 15*T1 - T1@T3
        gemm_wmma_kernel<false, 1, false><<<dim3(M_ / 128, M_ / 64, BH_), 128, 0, stream>>>(
            T1, T3, T1, T5, M_, M_, M_, M_, M_, sMM, sMM, sMM, sMM, -1.f, 15.f);
        // z' = 0.25*z@(13I - T5) = 3.25*z - 0.25*z@T5
        gemm_wmma_kernel<false, 1, false><<<dim3(M_ / 128, M_ / 64, BH_), 128, 0, stream>>>(
            zc, T5, zc, zn, M_, M_, M_, M_, M_, sMM, sMM, sMM, sMM, -0.25f, 3.25f);
        bf16_t* t = zc; zc = zn; zn = t;
    }

    // ---- 5) sim1 = q @ k_l^T -> softmax -> attn1 ----
    gemm_wmma_kernel<true, 0, true><<<dim3(N_ / 128, M_ / 64, BH_), 128, 0, stream>>>(
        q, k_l, nullptr, simf, DH_, DH_, DH_, 0, M_, sND, sMD, 0, sNM, 1.f, 0.f);
    softmax_rows_kernel<<<BH_ * N_, 256, 0, stream>>>(simf, A1, M_);

    // ---- 6) sim3 = q_l @ k^T -> softmax -> attn3 ----
    gemm_wmma_kernel<true, 0, true><<<dim3(M_ / 128, N_ / 64, BH_), 128, 0, stream>>>(
        q_l, k, nullptr, simf, DH_, DH_, DH_, 0, N_, sMD, sND, 0, sNM, 1.f, 0.f);
    softmax_rows_kernel<<<BH_ * M_, 256, 0, stream>>>(simf, A3, N_);

    // ---- 7) a3v = attn3 @ v   [256 x 64], K=4096 (NN) ----
    gemm_wmma_kernel<false, 0, false><<<dim3(M_ / 128, DH_ / 64, BH_), 128, 0, stream>>>(
        A3, v, nullptr, a3v, N_, N_, DH_, 0, DH_, sNM, sND, 0, sMD, 1.f, 0.f);

    // ---- 8) tA = attn1 @ z    [4096 x 256], K=256 (NN) ----
    bf16_t* tA = (bf16_t*)R1;  // sim region is free now
    gemm_wmma_kernel<false, 0, false><<<dim3(N_ / 128, M_ / 64, BH_), 128, 0, stream>>>(
        A1, zc, nullptr, tA, M_, M_, M_, 0, M_, sNM, sMM, 0, sNM, 1.f, 0.f);

    // ---- 9) outh = tA @ a3v   [4096 x 64], K=256 (NN); reuse q buffer ----
    bf16_t* outh = q;
    gemm_wmma_kernel<false, 0, false><<<dim3(N_ / 128, DH_ / 64, BH_), 128, 0, stream>>>(
        tA, a3v, nullptr, outh, M_, M_, DH_, 0, DH_, sNM, sMD, 0, sND, 1.f, 0.f);

    // ---- 10) merge heads -> xo (reuse k buffer), final projection + bias ----
    bf16_t* xo = k;
    merge_heads_kernel<<<(unsigned)(XE / 256), 256, 0, stream>>>(outh, xo);
    gemm_wmma_kernel<true, 2, true><<<dim3((B_ * N_) / 128, D_ / 64, 1), 128, 0, stream>>>(
        xo, woutT, bout, out, D_, D_, D_, 0, D_, 0, 0, 0, 0, 1.f, 0.f);
}
